// MultiHeadSelfAttention_23922967838958
// MI455X (gfx1250) — compile-verified, matrix operand order checked by reading, not run on hardware
//
#include <hip/hip_runtime.h>

typedef _Float16 v4h  __attribute__((ext_vector_type(4)));
typedef _Float16 v8h  __attribute__((ext_vector_type(8)));
typedef _Float16 v16h __attribute__((ext_vector_type(16)));
typedef float    v8f  __attribute__((ext_vector_type(8)));

#define BATCH   4
#define SEQ     2048
#define DMODEL  1024
#define NH      16
#define HD      64

// ---------------------------------------------------------------------------
// Fragment loader for WMMA f16 16x16x32, TN-style data (rows contiguous in K).
// Per CDNA5 ISA: lane l<16 holds row l, K = {0..7, 16..23};
// lane l>=16 holds row l-16, K = {8..15, 24..31}.  Works for both A fragments
// (row = M index) and B fragments (row = N index when B^T is row-major in K).
// ---------------------------------------------------------------------------
__device__ __forceinline__ v16h load_frag_g(const _Float16* __restrict__ base,
                                            int row, int ld, int k0) {
    const int lane = threadIdx.x & 31;
    const int half = lane >> 4;
    const _Float16* p = base + (size_t)row * ld + k0 + half * 8;
    v16h r;
    *(v8h*)&r       = *(const v8h*)(p);       // K = k0 + half*8 .. +7
    *((v8h*)&r + 1) = *(const v8h*)(p + 16);  // K = k0 + 16 + half*8 .. +7
    return r;
}

__device__ __forceinline__ v8f wmma_f16(v16h a, v16h b, v8f c) {
    return __builtin_amdgcn_wmma_f32_16x16x32_f16(false, a, false, b,
                                                  (short)0, c, false, false);
}

// ---------------------------------------------------------------------------
// fp32 -> fp16 conversion, 4 elements per thread
// ---------------------------------------------------------------------------
__global__ void cvt_f32_f16_x4(const float* __restrict__ in,
                               _Float16* __restrict__ out, int n4) {
    int i = blockIdx.x * blockDim.x + threadIdx.x;
    if (i < n4) {
        float4 v = ((const float4*)in)[i];
        v4h o = { (_Float16)v.x, (_Float16)v.y, (_Float16)v.z, (_Float16)v.w };
        ((v4h*)out)[i] = o;
    }
}

// ---------------------------------------------------------------------------
// QKV projection: [8192,1024] x [3072,1024]^T + bias.
// 256 threads = 8 waves; block tile 64(M) x 256(N); wave tile 32 x 64.
// Epilogue scatters to Q/K [B,H,S,64] (f16) and V transposed [B,H,64,S].
// ---------------------------------------------------------------------------
__global__ void qkv_gemm_wmma(const _Float16* __restrict__ xh,
                              const _Float16* __restrict__ wh,
                              const float*    __restrict__ bqkv,
                              _Float16* __restrict__ Qh,
                              _Float16* __restrict__ Kh,
                              _Float16* __restrict__ Vt) {
    const int lane = threadIdx.x & 31;
    const int wv   = threadIdx.x >> 5;
    const int half = lane >> 4;
    const int ln   = lane & 15;
    const int m0 = blockIdx.y * 64 + (wv & 1) * 32;
    const int n0 = blockIdx.x * 256 + (wv >> 1) * 64;

    v8f acc[2][4] = {};
    for (int k = 0; k < DMODEL; k += 32) {
        v16h a0 = load_frag_g(xh, m0 + ln,      DMODEL, k);
        v16h a1 = load_frag_g(xh, m0 + 16 + ln, DMODEL, k);
#pragma unroll
        for (int nf = 0; nf < 4; ++nf) {
            v16h b = load_frag_g(wh, n0 + nf * 16 + ln, DMODEL, k);
            acc[0][nf] = wmma_f16(a0, b, acc[0][nf]);
            acc[1][nf] = wmma_f16(a1, b, acc[1][nf]);
        }
    }
#pragma unroll
    for (int mf = 0; mf < 2; ++mf)
#pragma unroll
    for (int nf = 0; nf < 4; ++nf) {
        const int n = n0 + nf * 16 + ln;          // 0..3071 = which*1024+h*64+d
        const float bias = bqkv[n];
        const int which = n >> 10;
        const int h = (n & 1023) >> 6;
        const int d = n & 63;
#pragma unroll
        for (int r = 0; r < 8; ++r) {
            const int m = m0 + mf * 16 + r + half * 8;
            const int b_ = m / SEQ, s = m % SEQ;
            const size_t bh = (size_t)(b_ * NH + h);
            const _Float16 val = (_Float16)(acc[mf][nf][r] + bias);
            if (which == 0)      Qh[(bh * SEQ + s) * HD + d] = val;
            else if (which == 1) Kh[(bh * SEQ + s) * HD + d] = val;
            else                 Vt[(bh * HD + d) * SEQ + s] = val;
        }
    }
}

// ---------------------------------------------------------------------------
// Causal flash attention: 128 threads = 4 waves, 16 query rows per wave,
// key tiles of 32. Online softmax; P reshaped C->A layout through LDS.
// ---------------------------------------------------------------------------
__global__ void attn_flash_wmma(const _Float16* __restrict__ Qh,
                                const _Float16* __restrict__ Kh,
                                const _Float16* __restrict__ Vt,
                                _Float16* __restrict__ ctx) {
    __shared__ _Float16 psm[4][16][32];
    const int lane = threadIdx.x & 31;
    const int wv   = threadIdx.x >> 5;
    const int half = lane >> 4;
    const int ln   = lane & 15;

    const int qt = blockIdx.x % (SEQ / 64);
    const int bh = blockIdx.x / (SEQ / 64);
    const int h  = bh % NH;
    const int b  = bh / NH;
    const int q0 = qt * 64 + wv * 16;

    const _Float16* Qb = Qh + (size_t)bh * SEQ * HD;
    const _Float16* Kb = Kh + (size_t)bh * SEQ * HD;
    const _Float16* Vb = Vt + (size_t)bh * HD * SEQ;

    const v16h qa0 = load_frag_g(Qb, q0 + ln, HD, 0);
    const v16h qa1 = load_frag_g(Qb, q0 + ln, HD, 32);

    v8f o[4] = {};
    float mrun[8], lrun[8];
#pragma unroll
    for (int r = 0; r < 8; ++r) { mrun[r] = -1e30f; lrun[r] = 0.0f; }
    const float scale = 0.125f;  // 1/sqrt(64)

    const int kend = q0 + 16;    // causal: keys 0 .. q0+15
    for (int kt = 0; kt < kend; kt += 32) {
        // ---- scores S = Q K^T (16 x 32) ----
        v8f s0 = {}, s1 = {};
        v16h bk;
        bk = load_frag_g(Kb, kt + ln,      HD, 0);  s0 = wmma_f16(qa0, bk, s0);
        bk = load_frag_g(Kb, kt + ln,      HD, 32); s0 = wmma_f16(qa1, bk, s0);
        bk = load_frag_g(Kb, kt + 16 + ln, HD, 0);  s1 = wmma_f16(qa0, bk, s1);
        bk = load_frag_g(Kb, kt + 16 + ln, HD, 32); s1 = wmma_f16(qa1, bk, s1);

        // ---- mask + online softmax (row m = r + 8*half, col = ln) ----
        float p0[8], p1[8];
#pragma unroll
        for (int r = 0; r < 8; ++r) {
            const int qg = q0 + r + half * 8;
            float v0 = (kt + ln      <= qg) ? s0[r] * scale : -1e30f;
            float v1 = (kt + 16 + ln <= qg) ? s1[r] * scale : -1e30f;
            float mx = fmaxf(v0, v1);
            mx = fmaxf(mx, __shfl_xor(mx, 1, 16));
            mx = fmaxf(mx, __shfl_xor(mx, 2, 16));
            mx = fmaxf(mx, __shfl_xor(mx, 4, 16));
            mx = fmaxf(mx, __shfl_xor(mx, 8, 16));
            const float mn = fmaxf(mrun[r], mx);
            const float al = __expf(mrun[r] - mn);
            mrun[r] = mn;
            const float e0 = __expf(v0 - mn);
            const float e1 = __expf(v1 - mn);
            p0[r] = e0; p1[r] = e1;
            float rs = e0 + e1;
            rs += __shfl_xor(rs, 1, 16);
            rs += __shfl_xor(rs, 2, 16);
            rs += __shfl_xor(rs, 4, 16);
            rs += __shfl_xor(rs, 8, 16);
            lrun[r] = lrun[r] * al + rs;
            o[0][r] *= al; o[1][r] *= al; o[2][r] *= al; o[3][r] *= al;
        }

        // ---- C-layout P -> LDS -> A-layout fragment (wave-private tile) ----
#pragma unroll
        for (int r = 0; r < 8; ++r) {
            psm[wv][r + half * 8][ln]      = (_Float16)p0[r];
            psm[wv][r + half * 8][16 + ln] = (_Float16)p1[r];
        }
        asm volatile("s_wait_dscnt 0" ::: "memory");
        v16h pa;
        *(v8h*)&pa       = *(const v8h*)&psm[wv][ln][half * 8];
        *((v8h*)&pa + 1) = *(const v8h*)&psm[wv][ln][16 + half * 8];

        // ---- O += P V  (V^T stored [d][s]: contiguous B fragments) ----
#pragma unroll
        for (int nf = 0; nf < 4; ++nf) {
            v16h bv = load_frag_g(Vb, nf * 16 + ln, SEQ, kt);
            o[nf] = wmma_f16(pa, bv, o[nf]);
        }
    }

    // ---- normalize and emit ctx [B,S,1024] (f16, K-contiguous for out proj)
#pragma unroll
    for (int r = 0; r < 8; ++r) {
        const int s = q0 + r + half * 8;
        const float inv = 1.0f / lrun[r];
#pragma unroll
        for (int nf = 0; nf < 4; ++nf)
            ctx[((size_t)b * SEQ + s) * DMODEL + h * HD + nf * 16 + ln] =
                (_Float16)(o[nf][r] * inv);
    }
}

// ---------------------------------------------------------------------------
// Output projection: ctx[8192,1024] x w_out[1024,1024]^T + b_out -> f32 out
// ---------------------------------------------------------------------------
__global__ void out_gemm_wmma(const _Float16* __restrict__ ch,
                              const _Float16* __restrict__ wh,
                              const float*    __restrict__ bout,
                              float* __restrict__ out) {
    const int lane = threadIdx.x & 31;
    const int wv   = threadIdx.x >> 5;
    const int half = lane >> 4;
    const int ln   = lane & 15;
    const int m0 = blockIdx.y * 64 + (wv & 1) * 32;
    const int n0 = blockIdx.x * 256 + (wv >> 1) * 64;

    v8f acc[2][4] = {};
    for (int k = 0; k < DMODEL; k += 32) {
        v16h a0 = load_frag_g(ch, m0 + ln,      DMODEL, k);
        v16h a1 = load_frag_g(ch, m0 + 16 + ln, DMODEL, k);
#pragma unroll
        for (int nf = 0; nf < 4; ++nf) {
            v16h b = load_frag_g(wh, n0 + nf * 16 + ln, DMODEL, k);
            acc[0][nf] = wmma_f16(a0, b, acc[0][nf]);
            acc[1][nf] = wmma_f16(a1, b, acc[1][nf]);
        }
    }
#pragma unroll
    for (int mf = 0; mf < 2; ++mf)
#pragma unroll
    for (int nf = 0; nf < 4; ++nf) {
        const int n = n0 + nf * 16 + ln;
        const float bias = bout[n];
#pragma unroll
        for (int r = 0; r < 8; ++r) {
            const int m = m0 + mf * 16 + r + half * 8;
            out[(size_t)m * DMODEL + n] = acc[mf][nf][r] + bias;
        }
    }
}

// ---------------------------------------------------------------------------
extern "C" void kernel_launch(void* const* d_in, const int* in_sizes, int n_in,
                              void* d_out, int out_size, void* d_ws, size_t ws_size,
                              hipStream_t stream) {
    const float* x     = (const float*)d_in[0];
    // d_in[1] = mask: causal, recomputed analytically in-kernel
    const float* w_qkv = (const float*)d_in[2];
    const float* b_qkv = (const float*)d_in[3];
    const float* w_out = (const float*)d_in[4];
    const float* b_out = (const float*)d_in[5];

    char* ws = (char*)d_ws;
    size_t off = 0;
    auto alloc = [&](size_t bytes) {
        char* p = ws + off;
        off += (bytes + 255) & ~(size_t)255;
        return p;
    };
    _Float16* xh  = (_Float16*)alloc((size_t)BATCH * SEQ * DMODEL * 2);
    _Float16* wqh = (_Float16*)alloc((size_t)3 * DMODEL * DMODEL * 2);
    _Float16* woh = (_Float16*)alloc((size_t)DMODEL * DMODEL * 2);
    _Float16* Qh  = (_Float16*)alloc((size_t)BATCH * NH * SEQ * HD * 2);
    _Float16* Kh  = (_Float16*)alloc((size_t)BATCH * NH * SEQ * HD * 2);
    _Float16* Vt  = (_Float16*)alloc((size_t)BATCH * NH * HD * SEQ * 2);
    _Float16* ctx = (_Float16*)alloc((size_t)BATCH * SEQ * DMODEL * 2);

    const int nx = BATCH * SEQ * DMODEL / 4;     // 2,097,152
    const int nw = 3 * DMODEL * DMODEL / 4;      //   786,432
    const int no = DMODEL * DMODEL / 4;          //   262,144
    cvt_f32_f16_x4<<<(nx + 255) / 256, 256, 0, stream>>>(x, xh, nx);
    cvt_f32_f16_x4<<<(nw + 255) / 256, 256, 0, stream>>>(w_qkv, wqh, nw);
    cvt_f32_f16_x4<<<(no + 255) / 256, 256, 0, stream>>>(w_out, woh, no);

    qkv_gemm_wmma<<<dim3(3 * DMODEL / 256, BATCH * SEQ / 64), 256, 0, stream>>>(
        xh, wqh, b_qkv, Qh, Kh, Vt);

    attn_flash_wmma<<<dim3(BATCH * NH * (SEQ / 64)), 128, 0, stream>>>(
        Qh, Kh, Vt, ctx);

    out_gemm_wmma<<<dim3(DMODEL / 256, BATCH * SEQ / 64), 256, 0, stream>>>(
        ctx, woh, b_out, (float*)d_out);
}
